// FastRCNN_18399639896537
// MI455X (gfx1250) — matrix operand order repaired and checked by reading, not compiled, not run on hardware
//
#include <hip/hip_runtime.h>

// ---------------- problem constants ----------------
#define HH   200
#define WW   200
#define CC   256
#define NROI 4096
#define POOLB 7
#define SP   14          // POOL * G
#define FEAT 12544       // 49 * 256
#define HID  256
#define NCLS 21
#define KSTEP 64
#define NKB  196         // FEAT / 64

typedef __attribute__((ext_vector_type(16))) __bf16 v16bf;
typedef __attribute__((ext_vector_type(8)))  float  v8f;

// ---------------------------------------------------------------------------
// Kernel A: W1 (FEAT x HID, f32 row-major) -> W1T (HID x FEAT, bf16)
// so B-fragment loads in the GEMM are contiguous along K.
// ---------------------------------------------------------------------------
__global__ __launch_bounds__(256)
void w1_transpose_bf16(const float* __restrict__ W1, __bf16* __restrict__ W1T)
{
    int idx = blockIdx.x * 256 + threadIdx.x;   // n * FEAT + k, k fastest
    if (idx >= HID * FEAT) return;
    int n = idx / FEAT;
    int k = idx - n * FEAT;
    W1T[idx] = (__bf16)W1[(size_t)k * HID + n];
}

// ---------------------------------------------------------------------------
// Kernel B: fused ROI max-pool + GEMM1 + bias + ReLU.
//   grid.x = NROI/16 blocks, 512 threads = 16 waves.
//   Wave w computes the 16x16 tile (rows m0..m0+15, cols 16w..16w+15) of
//   H = relu(x @ W1 + b1); x is pooled on the fly into LDS (double buffered),
//   two v_wmma_f32_16x16x32_bf16 per barrier (K-step 64).
// ---------------------------------------------------------------------------
__global__ __launch_bounds__(512)
void gemm1_fused(const float* __restrict__ feat,
                 const float* __restrict__ rois,
                 const __bf16* __restrict__ w1t,
                 const float* __restrict__ b1,
                 float* __restrict__ Hout)
{
    __shared__ __bf16 As[2][16][72];   // 16 rows x 64 k (pad 72 -> 144B rows, 16B aligned)
    __shared__ int siy[16][SP];
    __shared__ int six[16][SP];

    const int tid = threadIdx.x;
    const int m0  = blockIdx.x * 16;

    // ---- per-ROI sampling index tables (rintf = round-half-even, matches jnp.round) ----
    if (tid < 16 * SP) {
        int r = tid / SP, j = tid % SP;
        float r0 = rois[(m0 + r) * 4 + 0];
        float r1 = rois[(m0 + r) * 4 + 1];
        float r2 = rois[(m0 + r) * 4 + 2];
        float r3 = rois[(m0 + r) * 4 + 3];
        float y1 = fminf(r0, r2), y2 = fmaxf(r0, r2);
        float x1 = fminf(r1, r3), x2 = fmaxf(r1, r3);
        float t  = ((float)j + 0.5f) / (float)SP;
        float ys = y1 + (y2 - y1) * t;
        float xs = x1 + (x2 - x1) * t;
        int iy = (int)rintf(ys * (float)(HH - 1));
        int ix = (int)rintf(xs * (float)(WW - 1));
        iy = min(max(iy, 0), HH - 1);
        ix = min(max(ix, 0), WW - 1);
        siy[r][j] = iy;
        six[r][j] = ix;
    }
    __syncthreads();

    const int r = tid >> 5;   // staging row 0..15 (ROI within tile)
    const int e = tid & 31;   // staging k-in-chunk (handles e and e+32)

    auto stage = [&](int kb, int buf) {
        int bin = kb >> 2;                 // 0..48  (each 64-chunk stays in one bin)
        int c   = ((kb & 3) << 6) + e;     // channel of first element
        int py  = bin / POOLB, px = bin % POOLB;
        int iy0 = siy[r][py * 2], iy1 = siy[r][py * 2 + 1];
        int ix0 = six[r][px * 2], ix1 = six[r][px * 2 + 1];
        size_t p00 = (size_t)(iy0 * WW + ix0) * CC + c;
        size_t p01 = (size_t)(iy0 * WW + ix1) * CC + c;
        size_t p10 = (size_t)(iy1 * WW + ix0) * CC + c;
        size_t p11 = (size_t)(iy1 * WW + ix1) * CC + c;
        float a0 = fmaxf(fmaxf(feat[p00],      feat[p01]),
                         fmaxf(feat[p10],      feat[p11]));
        float a1 = fmaxf(fmaxf(feat[p00 + 32], feat[p01 + 32]),
                         fmaxf(feat[p10 + 32], feat[p11 + 32]));
        As[buf][r][e]      = (__bf16)a0;
        As[buf][r][e + 32] = (__bf16)a1;
    };

    const int lane = tid & 31;
    const int wave = tid >> 5;          // 0..15 -> N tile
    const int n0   = wave * 16;
    const int lm   = lane & 15;         // matrix row (A) / column (B,C)
    const int hk   = lane >> 4;         // K half selector

    const float bias = b1[n0 + lm];

    v8f acc = {0.f, 0.f, 0.f, 0.f, 0.f, 0.f, 0.f, 0.f};

    stage(0, 0);
    __syncthreads();

    for (int kb = 0; kb < NKB; ++kb) {
        const int buf = kb & 1;
        if (kb + 1 < NKB) stage(kb + 1, buf ^ 1);

        // A fragments (two 16x32 bf16 tiles) from LDS:
        //   lanes 0-15: K 0-15 ; lanes 16-31: K 16-31
        v16bf a0, a1;
        const __bf16* arow = &As[buf][lm][0];
        #pragma unroll
        for (int i = 0; i < 8; ++i) {
            a0[i]     = arow[hk * 16 + i];
            a0[8 + i] = arow[hk * 16 + 8 + i];
            a1[i]     = arow[32 + hk * 16 + i];
            a1[8 + i] = arow[32 + hk * 16 + 8 + i];
        }

        // B fragments (32x16 bf16): lane = column n0+lm; K contiguous per half-wave
        v16bf b0, b1v;
        const __bf16* brow = w1t + (size_t)(n0 + lm) * FEAT + kb * KSTEP + hk * 16;
        #pragma unroll
        for (int i = 0; i < 16; ++i) {
            b0[i]  = brow[i];
            b1v[i] = brow[32 + i];
        }

        if (kb + 2 < NKB)
            __builtin_prefetch(w1t + (size_t)(n0 + lm) * FEAT + (kb + 2) * KSTEP, 0, 3);

        acc = __builtin_amdgcn_wmma_f32_16x16x32_bf16(
                  false, a0, false, b0,  (short)0, acc, false, false);
        acc = __builtin_amdgcn_wmma_f32_16x16x32_bf16(
                  false, a1, false, b1v, (short)0, acc, false, false);

        __syncthreads();   // stage(kb+1) complete; next iter overwrites buf kb&1
    }

    // C/D layout: reg i, lane L -> M = (L>>4)*8 + i, N = L&15
    #pragma unroll
    for (int i = 0; i < 8; ++i) {
        int mrow = m0 + hk * 8 + i;
        int ncol = n0 + lm;
        float v = acc[i] + bias;
        Hout[(size_t)mrow * HID + ncol] = fmaxf(v, 0.f);
    }
}

// ---------------------------------------------------------------------------
// Kernel C: heads + per-ROI loss. Wave per ROI (8 waves / block).
//   lanes 0..20  : class logits
//   lanes 24..27 : bbox_pred for the ROI's label
// ---------------------------------------------------------------------------
__global__ __launch_bounds__(256)
void head_loss(const float* __restrict__ Hm,
               const float* __restrict__ Wc, const float* __restrict__ bc,
               const float* __restrict__ Wb, const float* __restrict__ bb,
               const int*   __restrict__ labels,
               const float* __restrict__ tgt,
               float* __restrict__ Lout)
{
    __shared__ float hs[8][HID];
    const int t    = threadIdx.x;
    const int roi0 = blockIdx.x * 8;

    #pragma unroll
    for (int i = 0; i < 8; ++i)
        hs[i][t] = Hm[(size_t)(roi0 + i) * HID + t];
    __syncthreads();

    const int w = t >> 5, l = t & 31;
    const int roi = roi0 + w;
    const int label = labels[roi];

    float logit = 0.f, bpred = 0.f;
    if (l < NCLS) {
        for (int c = 0; c < HID; ++c)
            logit += hs[w][c] * Wc[c * NCLS + l];
        logit += bc[l];
    } else if (l >= 24 && l < 28) {
        const int j = l - 24;
        for (int c = 0; c < HID; ++c)
            bpred += hs[w][c] * Wb[c * (NCLS * 4) + label * 4 + j];
        bpred += bb[label * 4 + j];
    }

    // log-sum-exp over 21 logits via wave32 shuffles
    float m = -3.402823466e38f;
    for (int j = 0; j < NCLS; ++j)
        m = fmaxf(m, __shfl(logit, j, 32));
    float se = 0.f;
    for (int j = 0; j < NCLS; ++j)
        se += expf(__shfl(logit, j, 32) - m);
    float lse = logf(se) + m;
    float cls = lse - __shfl(logit, label, 32);

    float reg = 0.f;
    for (int j = 0; j < 4; ++j) {
        float d  = __shfl(bpred, 24 + j, 32) - tgt[roi * 4 + j];
        float ad = fabsf(d);
        reg += (ad < 1.f) ? 0.5f * d * d : (ad - 0.5f);
    }

    if (l == 0) Lout[roi] = cls + reg;
}

// ---------------------------------------------------------------------------
// Kernel D: deterministic mean reduction to scalar.
// ---------------------------------------------------------------------------
__global__ __launch_bounds__(256)
void reduce_mean(const float* __restrict__ Lout, float* __restrict__ out)
{
    __shared__ float s[256];
    const int t = threadIdx.x;
    float acc = 0.f;
    for (int i = t; i < NROI; i += 256) acc += Lout[i];
    s[t] = acc;
    __syncthreads();
    for (int off = 128; off > 0; off >>= 1) {
        if (t < off) s[t] += s[t + off];
        __syncthreads();
    }
    if (t == 0) out[0] = s[0] / (float)NROI;
}

// ---------------------------------------------------------------------------
extern "C" void kernel_launch(void* const* d_in, const int* in_sizes, int n_in,
                              void* d_out, int out_size, void* d_ws, size_t ws_size,
                              hipStream_t stream)
{
    const float* features = (const float*)d_in[0];
    const float* rois     = (const float*)d_in[1];
    const int*   labels   = (const int*)  d_in[2];
    const float* tgts     = (const float*)d_in[3];
    const float* W1       = (const float*)d_in[4];
    const float* b1       = (const float*)d_in[5];
    const float* Wc       = (const float*)d_in[6];
    const float* bc       = (const float*)d_in[7];
    const float* Wb       = (const float*)d_in[8];
    const float* bb       = (const float*)d_in[9];
    float* out = (float*)d_out;

    char* ws = (char*)d_ws;
    size_t szW1T = (size_t)HID * FEAT * sizeof(__bf16);            // ~6.4 MB
    size_t off1  = (szW1T + 255) & ~(size_t)255;
    size_t szH   = (size_t)NROI * HID * sizeof(float);             // 4 MB
    size_t off2  = (off1 + szH + 255) & ~(size_t)255;

    __bf16* W1T = (__bf16*)ws;
    float*  Hm  = (float*)(ws + off1);
    float*  Lo  = (float*)(ws + off2);

    w1_transpose_bf16<<<(HID * FEAT + 255) / 256, 256, 0, stream>>>(W1, W1T);
    gemm1_fused<<<NROI / 16, 512, 0, stream>>>(features, rois, W1T, b1, Hm);
    head_loss<<<NROI / 8, 256, 0, stream>>>(Hm, Wc, bc, Wb, bb, labels, tgts, Lo);
    reduce_mean<<<1, 256, 0, stream>>>(Lo, out);
}
